// Self_Attn_37641093382312
// MI455X (gfx1250) — matrix operand order repaired and checked by reading, not compiled, not run on hardware
//
#include <hip/hip_runtime.h>

// ---------------------------------------------------------------------------
// Self-attention (SAGAN) for MI455X gfx1250: bf16 WMMA, flash-style softmax.
// B=8, C=1024, N=64*64=4096, C_ATTN=32.
// Pipeline: cvt weights -> transpose+cvt x -> QK proj -> V proj -> softmax
// stats (two-pass) -> fused attention (double-buffered beta tiles, one
// barrier per 128-j superstep). All GEMM inner loops are conversion-free
// with contiguous fragment loads (k = 16*half + s in A and B alike).
// ---------------------------------------------------------------------------

#define NPIX 4096
#define CIN  1024
#define CATT 32
#define NB   8

typedef __attribute__((ext_vector_type(16))) __bf16          v16bf;
typedef __attribute__((ext_vector_type(16))) unsigned short  v16u;
typedef __attribute__((ext_vector_type(8)))  unsigned short  u8x;
typedef __attribute__((ext_vector_type(4)))  unsigned short  u4s;
typedef __attribute__((ext_vector_type(8)))  float           v8f;
typedef __attribute__((ext_vector_type(4)))  float           f4x;

// Hardware f32 -> bf16 (RNE) via native conversion.
__device__ __forceinline__ unsigned short f2bf(float f) {
  return __builtin_bit_cast(unsigned short, (__bf16)f);
}

__device__ __forceinline__ v16bf bfc(v16u u) {
  return __builtin_bit_cast(v16bf, u);
}

// Assemble an A-fragment (16x32 bf16, MxK) register image from the two
// contiguous 8-element halves (16 consecutive k; callers use k = 16*half + s, as B does).
__device__ __forceinline__ v16bf mkA(const unsigned short* lo_p) {
  const u8x lo = *(const u8x*)lo_p;
  const u8x hi = *(const u8x*)(lo_p + 8);
  v16u u;
#pragma unroll
  for (int t = 0; t < 8; ++t) { u[t] = lo[t]; u[t + 8] = hi[t]; }
  return __builtin_bit_cast(v16bf, u);
}

__device__ __forceinline__ v8f wmma_bf16(v16bf a, v16bf b, v8f c) {
  return __builtin_amdgcn_wmma_f32_16x16x32_bf16(
      /*neg_a=*/false, a, /*neg_b=*/false, b,
      /*c_mod=*/(short)0, c, /*reuse_a=*/false, /*reuse_b=*/false);
}

// ---------------------------------------------------------------------------
// Kernel 0a: one float4 per thread fp32 -> bf16 conversion (weights).
// ---------------------------------------------------------------------------
__global__ __launch_bounds__(256) void cvt_bf16_kernel(
    const float* __restrict__ in, unsigned short* __restrict__ outp, int n4) {
  const int i = blockIdx.x * 256 + threadIdx.x;
  if (i < n4) {
    const f4x v = ((const f4x*)in)[i];
    u4s o;
#pragma unroll
    for (int t = 0; t < 4; ++t) o[t] = f2bf(v[t]);
    ((u4s*)outp)[i] = o;
  }
}

// ---------------------------------------------------------------------------
// Kernel 0b: xT[b,n,c] = bf16(x[b,c,n]). Tiled 64x64 via LDS so both the
// global read and the global write are fully coalesced.
// ---------------------------------------------------------------------------
__global__ __launch_bounds__(256) void xpose_kernel(
    const float* __restrict__ x, unsigned short* __restrict__ xT) {
  __shared__ unsigned short tile[64 * 66];
  const int blk = blockIdx.x;                  // b*1024 + ct*64 + nt
  const int b = blk >> 10, ct = (blk >> 6) & 15, nt = blk & 63;
  const int c0 = ct << 6, n0 = nt << 6;
#pragma unroll
  for (int p = 0; p < 16; ++p) {
    const int idx = threadIdx.x + p * 256;
    const int cl = idx >> 6, nl = idx & 63;
    tile[cl * 66 + nl] = f2bf(x[((size_t)b * CIN + c0 + cl) * NPIX + n0 + nl]);
  }
  __syncthreads();
#pragma unroll
  for (int p = 0; p < 16; ++p) {
    const int idx = threadIdx.x + p * 256;
    const int nr = idx >> 6, cc = idx & 63;
    xT[((size_t)b * NPIX + n0 + nr) * CIN + c0 + cc] = tile[cc * 66 + nr];
  }
}

// ---------------------------------------------------------------------------
// Kernel 1a: Q[b,n,32] = (Wg x)^T + bg ; K[b,n,32] = (Wf x)^T + bf  (bf16)
// One wave per 16-pixel tile. A = xT rows (n x c), B = W rows (c x o).
// ---------------------------------------------------------------------------
__global__ __launch_bounds__(256) void proj_qk_kernel(
    const unsigned short* __restrict__ xT,
    const unsigned short* __restrict__ Wf16, const float* __restrict__ bfv,
    const unsigned short* __restrict__ Wg16, const float* __restrict__ bgv,
    unsigned short* __restrict__ Qb, unsigned short* __restrict__ Kb) {
  const int lane = threadIdx.x & 31, wv = threadIdx.x >> 5;
  const int gw  = blockIdx.x * 8 + wv;        // 2048 waves
  const int b   = gw >> 8;
  const int n0  = (gw & 255) << 4;
  const int grp = lane >> 4, ln = lane & 15;
  const unsigned short* xr = xT + ((size_t)b * NPIX + n0 + ln) * CIN;

  v8f q0 = {}, q1 = {}, k0 = {}, k1 = {};
  for (int c0 = 0; c0 < CIN; c0 += 32) {
    const v16bf a = mkA(xr + c0 + 16 * grp);
    const v16bf bg0 = bfc(*(const v16u*)(Wg16 + (size_t)ln * CIN + c0 + 16 * grp));
    const v16bf bg1 = bfc(*(const v16u*)(Wg16 + (size_t)(ln + 16) * CIN + c0 + 16 * grp));
    const v16bf bk0 = bfc(*(const v16u*)(Wf16 + (size_t)ln * CIN + c0 + 16 * grp));
    const v16bf bk1 = bfc(*(const v16u*)(Wf16 + (size_t)(ln + 16) * CIN + c0 + 16 * grp));
    q0 = wmma_bf16(a, bg0, q0);
    q1 = wmma_bf16(a, bg1, q1);
    k0 = wmma_bf16(a, bk0, k0);
    k1 = wmma_bf16(a, bk1, k1);
  }
  const float bq0 = bgv[ln], bq1 = bgv[ln + 16];
  const float bk0s = bfv[ln], bk1s = bfv[ln + 16];
#pragma unroll
  for (int r = 0; r < 8; ++r) {
    const int n = n0 + r + 8 * grp;
    const size_t base = ((size_t)b * NPIX + n) * CATT;
    Qb[base + ln]      = f2bf(q0[r] + bq0);
    Qb[base + 16 + ln] = f2bf(q1[r] + bq1);
    Kb[base + ln]      = f2bf(k0[r] + bk0s);
    Kb[base + 16 + ln] = f2bf(k1[r] + bk1s);
  }
}

// ---------------------------------------------------------------------------
// Kernel 1b: V[b,c,n] = Wh x + bh (bf16, channel-major so the o-GEMM uses V
// as A directly). Wave tile = 64o x 64n (16 accumulators).
// ---------------------------------------------------------------------------
__global__ __launch_bounds__(256) void proj_v_kernel(
    const unsigned short* __restrict__ xT, const unsigned short* __restrict__ Wh16,
    const float* __restrict__ bhv, unsigned short* __restrict__ Vb) {
  const int lane = threadIdx.x & 31, wv = threadIdx.x >> 5;
  const int gw  = blockIdx.x * 8 + wv;        // 8192 waves
  const int b   = gw >> 10;
  const int rem = gw & 1023;
  const int o0  = (rem >> 6) << 6;            // 16 o-blocks of 64
  const int n0  = (rem & 63) << 6;            // 64 n-blocks of 64
  const int grp = lane >> 4, ln = lane & 15;

  v8f acc[4][4] = {};                         // [o-sub][n-sub]
  for (int c0 = 0; c0 < CIN; c0 += 32) {
    v16bf av[4];
#pragma unroll
    for (int of = 0; of < 4; ++of)
      av[of] = mkA(Wh16 + (size_t)(o0 + of * 16 + ln) * CIN + c0 + 16 * grp);
#pragma unroll
    for (int ns = 0; ns < 4; ++ns) {
      const v16bf bx = bfc(*(const v16u*)(
          xT + ((size_t)b * NPIX + n0 + ns * 16 + ln) * CIN + c0 + 16 * grp));
#pragma unroll
      for (int of = 0; of < 4; ++of)
        acc[of][ns] = wmma_bf16(av[of], bx, acc[of][ns]);
    }
  }
#pragma unroll
  for (int of = 0; of < 4; ++of) {
#pragma unroll
    for (int r = 0; r < 8; ++r) {
      const int oc = o0 + of * 16 + r + 8 * grp;
      const float bias = bhv[oc];
      const size_t base = ((size_t)b * CIN + oc) * NPIX + n0;
#pragma unroll
      for (int ns = 0; ns < 4; ++ns)
        Vb[base + ns * 16 + ln] = f2bf(acc[of][ns][r] + bias);
    }
  }
}

// ---------------------------------------------------------------------------
// Kernel 2: two-pass softmax stats. Pass 1: m = max_j s (WMMA + fmax only).
// Pass 2: l = sum_j exp(s - m). Butterfly reductions across the 16 lanes of
// each half-wave holding a tile row. Recomputing the 8.6 GF score GEMM is
// far cheaper than the online-softmax VALU chain.
// ---------------------------------------------------------------------------
__global__ __launch_bounds__(256) void stats_kernel(
    const unsigned short* __restrict__ Qb, const unsigned short* __restrict__ Kb,
    float* __restrict__ mOut, float* __restrict__ lOut) {
  const int lane = threadIdx.x & 31, wv = threadIdx.x >> 5;
  const int gw = blockIdx.x * 8 + wv;
  const int b = gw >> 8, i0 = (gw & 255) << 4;
  const int grp = lane >> 4, ln = lane & 15;
  const unsigned short* Qbb = Qb + (size_t)b * NPIX * CATT;
  const unsigned short* Kbb = Kb + (size_t)b * NPIX * CATT;

  const v16bf aq = mkA(Qbb + (size_t)(i0 + ln) * CATT + 16 * grp);

  // ---- pass 1: row max ----
  float rm[8];
#pragma unroll
  for (int r = 0; r < 8; ++r) rm[r] = -__builtin_inff();
  for (int j0 = 0; j0 < NPIX; j0 += 16) {
    const v16u bu = *(const v16u*)(Kbb + (size_t)(j0 + ln) * CATT + 16 * grp);
    v8f z = {};
    const v8f s = wmma_bf16(aq, bfc(bu), z);
#pragma unroll
    for (int r = 0; r < 8; ++r) rm[r] = fmaxf(rm[r], s[r]);
  }
#pragma unroll
  for (int mask = 1; mask < 16; mask <<= 1)
#pragma unroll
    for (int r = 0; r < 8; ++r) rm[r] = fmaxf(rm[r], __shfl_xor(rm[r], mask, 32));

  // ---- pass 2: sum of exp(s - m) ----
  float rs[8];
#pragma unroll
  for (int r = 0; r < 8; ++r) rs[r] = 0.f;
  for (int j0 = 0; j0 < NPIX; j0 += 16) {
    const v16u bu = *(const v16u*)(Kbb + (size_t)(j0 + ln) * CATT + 16 * grp);
    v8f z = {};
    const v8f s = wmma_bf16(aq, bfc(bu), z);
#pragma unroll
    for (int r = 0; r < 8; ++r) rs[r] += __expf(s[r] - rm[r]);
  }
#pragma unroll
  for (int mask = 1; mask < 16; mask <<= 1)
#pragma unroll
    for (int r = 0; r < 8; ++r) rs[r] += __shfl_xor(rs[r], mask, 32);

  if (ln == 0) {
#pragma unroll
    for (int r = 0; r < 8; ++r) {
      const int i = i0 + r + 8 * grp;
      mOut[b * NPIX + i] = rm[r];
      lOut[b * NPIX + i] = rs[r];
    }
  }
}

// ---------------------------------------------------------------------------
// Kernel 3: fused attention. Block = 512 thr (16 waves), one (b, 64-row
// i-tile) each. Beta tiles are double-buffered in LDS: in iteration js each
// wave produces its 32j x 16i exp(s-m) tile for superstep js+1 into the
// other buffer, then consumes superstep js with 64 accumulate WMMAs — a
// single barrier per 128-j superstep. Epilogue: out = gamma * o / l + x.
// ---------------------------------------------------------------------------
__global__ __launch_bounds__(512) void attn_kernel(
    const unsigned short* __restrict__ Qb, const unsigned short* __restrict__ Kb,
    const unsigned short* __restrict__ Vb,
    const float* __restrict__ mIn, const float* __restrict__ lIn,
    const float* __restrict__ x, const float* __restrict__ gptr,
    float* __restrict__ out) {
  __shared__ __align__(32) unsigned short betas[2 * 16 * 512];  // 32 KB
  const int lane = threadIdx.x & 31, w = threadIdx.x >> 5;
  const int b = blockIdx.x >> 6;
  const int i0 = (blockIdx.x & 63) << 6;
  const int grp = lane >> 4, ln = lane & 15;
  const int jb = w & 3, it = w >> 2;
  const unsigned short* Kbb = Kb + (size_t)b * NPIX * CATT;
  const unsigned short* Vbb = Vb + (size_t)b * CIN * NPIX;

  // Persistent Q B-fragment (o-dim x this wave's i subtile), k = 16*half + s as in the A loads.
  const int iq = i0 + it * 16 + ln;
  const v16bf bq = bfc(*(const v16u*)(Qb + ((size_t)b * NPIX + iq) * CATT + 16 * grp));
  const float mi = mIn[b * NPIX + iq];
  const int c0 = w << 6;

  v8f acc[4][4] = {};

  // Produce exp(s^T - m) tile for superstep js into buffer buf.
  auto produce = [&](int js, int buf) {
    const int jbase = js * 128 + jb * 32;
    const unsigned short* kr0 = Kbb + (size_t)(jbase + ln) * CATT + 16 * grp;
    const v16bf ak0 = mkA(kr0);
    const v16bf ak1 = mkA(kr0 + 16 * CATT);
    v8f z = {};
    const v8f s0 = wmma_bf16(ak0, bq, z);
    const v8f s1 = wmma_bf16(ak1, bq, z);
    unsigned short* mt = &betas[buf * 8192 + w * 512];
#pragma unroll
    for (int r = 0; r < 8; ++r) {
      const int jl = r + 8 * grp;
      mt[(ln << 4) + jl]        = f2bf(__expf(s0[r] - mi));
      mt[((ln + 16) << 4) + jl] = f2bf(__expf(s1[r] - mi));
    }
  };

  // Consume superstep js from buffer buf: 64 accumulate WMMAs.
  auto consume = [&](int js, int buf) {
#pragma unroll
    for (int jb2 = 0; jb2 < 4; ++jb2) {
      const int jg = js * 128 + jb2 * 32;
      v16bf bb[4];
#pragma unroll
      for (int it2 = 0; it2 < 4; ++it2)
        bb[it2] = bfc(*(const v16u*)&betas[buf * 8192 + (it2 * 4 + jb2) * 512 +
                                           ((ln + 16 * grp) << 4)]);
#pragma unroll
      for (int cf = 0; cf < 4; ++cf) {
        const v16bf av =
            mkA(Vbb + (size_t)(c0 + cf * 16 + ln) * NPIX + jg + 16 * grp);
#pragma unroll
        for (int it2 = 0; it2 < 4; ++it2)
          acc[cf][it2] = wmma_bf16(av, bb[it2], acc[cf][it2]);
      }
    }
  };

  produce(0, 0);
  __syncthreads();
  for (int js = 0; js < 32; ++js) {
    const int sel = js & 1;
    if (js < 31) produce(js + 1, sel ^ 1);
    consume(js, sel);
    __syncthreads();
  }

  // ---- epilogue: out = gamma * o / l + x ----
  const float gamma = *gptr;
#pragma unroll
  for (int it2 = 0; it2 < 4; ++it2) {
    const int i = i0 + it2 * 16 + ln;
    const float sc = gamma / lIn[b * NPIX + i];
#pragma unroll
    for (int cf = 0; cf < 4; ++cf) {
#pragma unroll
      for (int r = 0; r < 8; ++r) {
        const int c = c0 + cf * 16 + r + 8 * grp;
        const size_t idx = ((size_t)b * CIN + c) * NPIX + i;
        out[idx] = fmaf(acc[cf][it2][r], sc, x[idx]);
      }
    }
  }
}

// ---------------------------------------------------------------------------
extern "C" void kernel_launch(void* const* d_in, const int* in_sizes, int n_in,
                              void* d_out, int out_size, void* d_ws, size_t ws_size,
                              hipStream_t stream) {
  (void)in_sizes; (void)n_in; (void)out_size; (void)ws_size;
  const float* x   = (const float*)d_in[0];
  const float* Wf  = (const float*)d_in[1];
  const float* bfv = (const float*)d_in[2];
  const float* Wg  = (const float*)d_in[3];
  const float* bgv = (const float*)d_in[4];
  const float* Wh  = (const float*)d_in[5];
  const float* bhv = (const float*)d_in[6];
  const float* gpt = (const float*)d_in[7];
  float* out = (float*)d_out;

  char* ws = (char*)d_ws;
  const size_t xtBytes = (size_t)NB * NPIX * CIN * 2;       // 64 MB
  const size_t qBytes  = (size_t)NB * NPIX * CATT * 2;      // 2 MB each
  unsigned short* xTb  = (unsigned short*)ws;  ws += xtBytes;
  unsigned short* Qb   = (unsigned short*)ws;  ws += qBytes;
  unsigned short* Kb   = (unsigned short*)ws;  ws += qBytes;
  unsigned short* Vb   = (unsigned short*)ws;  ws += (size_t)NB * CIN * NPIX * 2;  // 64 MB
  unsigned short* Wg16 = (unsigned short*)ws;  ws += (size_t)CATT * CIN * 2;
  unsigned short* Wf16 = (unsigned short*)ws;  ws += (size_t)CATT * CIN * 2;
  unsigned short* Wh16 = (unsigned short*)ws;  ws += (size_t)CIN * CIN * 2;
  float* mBuf = (float*)ws;                    ws += (size_t)NB * NPIX * 4;
  float* lBuf = (float*)ws;

  // Prep: bf16 weights + transposed bf16 x.
  cvt_bf16_kernel<<<(CATT * CIN / 4 + 255) / 256, 256, 0, stream>>>(Wg, Wg16, CATT * CIN / 4);
  cvt_bf16_kernel<<<(CATT * CIN / 4 + 255) / 256, 256, 0, stream>>>(Wf, Wf16, CATT * CIN / 4);
  cvt_bf16_kernel<<<(CIN * CIN / 4 + 255) / 256, 256, 0, stream>>>(Wh, Wh16, CIN * CIN / 4);
  xpose_kernel<<<NB * 16 * 64, 256, 0, stream>>>(x, xTb);

  proj_qk_kernel<<<256, 256, 0, stream>>>(xTb, Wf16, bfv, Wg16, bgv, Qb, Kb);
  proj_v_kernel<<<1024, 256, 0, stream>>>(xTb, Wh16, bhv, Vb);
  stats_kernel<<<256, 256, 0, stream>>>(Qb, Kb, mBuf, lBuf);
  attn_kernel<<<512, 512, 0, stream>>>(Qb, Kb, Vb, mBuf, lBuf, x, gpt, out);
}
